// precalcSDE_36146444763417
// MI455X (gfx1250) — matrix operand order repaired and checked by reading, not compiled, not run on hardware
//
#include <hip/hip_runtime.h>
#include <hip/hip_bf16.h>

typedef __attribute__((ext_vector_type(16))) __bf16 v16bf;
typedef __attribute__((ext_vector_type(8)))  __bf16 v8bf;
typedef __attribute__((ext_vector_type(8)))  float  v8f;
typedef __attribute__((ext_vector_type(4)))  float  v4f;

#define V_WORD      32000
#define DIM         128
#define VW          144        // staged V row width: 128 d-cols + 16 ones-cols
#define LATENT      10000
#define LATENT_PAD  10016      // next multiple of 32; padded rows/cols are zero-filled
#define NSPECIAL    4
#define BT          (8 * 1024)
#define LOG2E       1.4426950408889634f

// ---------------------------------------------------------------------------
// Kernel 0: latent_mat fp32 -> bf16, in both orientations.
//   lat  [LATENT_PAD][DIM]  : B-operand of Q@K^T (lane = latent = N, elems = d = K)
//   latT [DIM][LATENT_PAD]  : B-operand of P@V (lane = d = N, elems = latent = K)
// Zero-fills padding so masked tail latents contribute exact zeros.
// ---------------------------------------------------------------------------
__global__ void prep_latent_kernel(const float* __restrict__ latent,
                                   __bf16* __restrict__ lat,
                                   __bf16* __restrict__ latT) {
    int gid = blockIdx.x * blockDim.x + threadIdx.x;
    if (gid >= LATENT_PAD * DIM) return;
    int l = gid / DIM, d = gid - l * DIM;
    float v = (l < LATENT) ? latent[(size_t)l * DIM + d] : 0.0f;
    __bf16 b = (__bf16)v;
    lat [(size_t)l * DIM + d]        = b;
    latT[(size_t)d * LATENT_PAD + l] = b;
}

// ---------------------------------------------------------------------------
// Kernel 1: EmbeddingBag(sum) + tanh.  One wave per word; lane owns 4 floats.
// Q is pre-scaled by LOG2E so softmax runs natively in the exp2 domain.
// ---------------------------------------------------------------------------
__global__ void bag_tanh_kernel(const float* __restrict__ ngram_emb,
                                const int*   __restrict__ ids,
                                const int*   __restrict__ offsets,
                                int n_total,
                                float*  __restrict__ lang,
                                __bf16* __restrict__ qb) {
    int word = blockIdx.x * (blockDim.x >> 5) + (threadIdx.x >> 5);
    int lane = threadIdx.x & 31;
    if (word >= V_WORD) return;
    int beg = offsets[word];
    int end = (word + 1 < V_WORD) ? offsets[word + 1] : n_total;
    v4f acc = {0.f, 0.f, 0.f, 0.f};
    for (int k = beg; k < end; ++k) {
        const v4f* row = (const v4f*)(ngram_emb + (size_t)ids[k] * DIM);
        acc += row[lane];
    }
    size_t base = (size_t)word * DIM + lane * 4;
    #pragma unroll
    for (int i = 0; i < 4; ++i) {
        float t = tanhf(acc[i]);
        lang[base + i] = t;
        qb[base + i]   = (__bf16)(t * LOG2E);   // scores directly in log2 domain
    }
}

// ---------------------------------------------------------------------------
// Kernel 2: fused flash attention over the latent codebook.
//   token_emb = lang_emb + softmax(lang_emb @ latT) @ lat
// 8 waves/block, one 16-row Q tile per wave; 32-latent chunks staged once per
// block into double-buffered LDS. Softmax denominator is produced by the
// matrix pipe itself: staged V carries 16 extra ones-columns and a 9th
// accumulator tile integrates rowsum(P) under the same online rescaling.
// ---------------------------------------------------------------------------
__launch_bounds__(256)
__global__ void latent_attn_kernel(const __bf16* __restrict__ qb,
                                   const __bf16* __restrict__ lat,
                                   const __bf16* __restrict__ latT,
                                   const float*  __restrict__ lang,
                                   float* __restrict__ token) {
    __shared__ __align__(16) __bf16 ldsKT[2][128][32];  // [d][latent]   (P@V B operand)
    __shared__ __align__(16) __bf16 ldsV [2][32][VW];   // [latent][d | ones]
    __shared__ __align__(16) __bf16 pbuf[8][16 * 32];

    const int tid     = threadIdx.x;
    const int wave    = tid >> 5;
    const int lane    = tid & 31;
    const int half    = lane >> 4;
    const int l16     = lane & 15;
    const int rowbase = (blockIdx.x * 8 + wave) * 16;

    // ---- one-time: fill the ones-block (cols 128..143 of every V row) ----
    for (int i = tid; i < 2 * 32 * 16; i += 256) {
        int bi = i >> 9, r = (i >> 4) & 31, c = i & 15;
        ldsV[bi][r][128 + c] = (__bf16)1.0f;
    }

    // ---- block-cooperative staging of one 32-latent chunk into LDS ----
    auto stage = [&](int bi, int nbase) {
        {   // latT chunk: 128 rows x 32 cols (64 B/row), 2 threads/row x 32 B
            int row = tid >> 1, seg = tid & 1;
            const v8bf* src = (const v8bf*)(latT + (size_t)row * LATENT_PAD + nbase + seg * 16);
            v8bf* dst = (v8bf*)(&ldsKT[bi][row][seg * 16]);
            dst[0] = src[0];
            dst[1] = src[1];
        }
        {   // lat chunk: 32 rows x 128 data cols (256 B/row), 8 threads/row x 32 B
            int row = tid >> 3, seg = tid & 7;
            const v8bf* src = (const v8bf*)(lat + (size_t)(nbase + row) * DIM + seg * 16);
            v8bf* dst = (v8bf*)(&ldsV[bi][row][seg * 16]);
            dst[0] = src[0];
            dst[1] = src[1];
        }
    };

    // ---- load Q tile as four 16x32 A-fragments (K = 0..127) ----
    v16bf qfrag[4];
    const __bf16* qrow = qb + (size_t)(rowbase + l16) * DIM;
    #pragma unroll
    for (int kc = 0; kc < 4; ++kc) {
        union { v16bf v; v8bf h[2]; } f;
        f.h[0] = *(const v8bf*)(qrow + kc * 32 + 8 * half);
        f.h[1] = *(const v8bf*)(qrow + kc * 32 + 16 + 8 * half);
        qfrag[kc] = f.v;
    }

    const v8f vzero = {0.f,0.f,0.f,0.f,0.f,0.f,0.f,0.f};
    v8f acc[9];                     // 8 O tiles + rowsum tile
    float m[8];
    #pragma unroll
    for (int t = 0; t < 9; ++t) acc[t] = vzero;
    #pragma unroll
    for (int v = 0; v < 8; ++v) m[v] = -1e30f;

    const int NITER = LATENT_PAD / 32;                  // 313
    stage(0, 0);

    for (int nb = 0; nb < NITER; ++nb) {
        const int cur   = nb & 1;
        const int nbase = nb * 32;
        __syncthreads();                                 // buf[cur] ready; prev reads done
        if (nb + 1 < NITER) stage(cur ^ 1, nbase + 32);  // overlaps with compute below
        if (nb + 2 < NITER)                              // warm L2 two chunks ahead
            __builtin_prefetch(lat + (size_t)(nbase + 64 + lane) * DIM, 0, 0);

        // ---- S = Q @ K^T : 8 WMMAs fed by bulk LDS loads ----
        v16bf bq[8];                                     // [nt][kc]
        #pragma unroll
        for (int nt = 0; nt < 2; ++nt)
            #pragma unroll
            for (int kc = 0; kc < 4; ++kc) {
                union { v16bf v; v8bf h[2]; } b;
                const __bf16* bp = &ldsV[cur][nt * 16 + l16][kc * 32 + 8 * half];
                b.h[0] = *(const v8bf*)(bp);
                b.h[1] = *(const v8bf*)(bp + 16);
                bq[nt * 4 + kc] = b.v;
            }
        v8f s[2];
        s[0] = vzero; s[1] = vzero;
        #pragma unroll
        for (int nt = 0; nt < 2; ++nt)
            #pragma unroll
            for (int kc = 0; kc < 4; ++kc)
                s[nt] = __builtin_amdgcn_wmma_f32_16x16x32_bf16(
                            false, qfrag[kc], false, bq[nt * 4 + kc],
                            (short)0, s[nt], false, false);

        // ---- tail masking only on the final chunk (uniform branch) ----
        if (nbase + 32 > LATENT) {
            #pragma unroll
            for (int nt = 0; nt < 2; ++nt) {
                const int col = nbase + nt * 16 + l16;
                #pragma unroll
                for (int v = 0; v < 8; ++v)
                    if (col >= LATENT) s[nt][v] = -1e30f;
            }
        }

        // ---- online softmax: row max via lane shuffles, rescale 9 tiles ----
        #pragma unroll
        for (int v = 0; v < 8; ++v) {
            float rm = fmaxf(s[0][v], s[1][v]);
            rm = fmaxf(rm, __shfl_xor(rm, 1));
            rm = fmaxf(rm, __shfl_xor(rm, 2));
            rm = fmaxf(rm, __shfl_xor(rm, 4));
            rm = fmaxf(rm, __shfl_xor(rm, 8));
            const float mn = fmaxf(m[v], rm);
            const float sc = exp2f(m[v] - mn);
            m[v] = mn;
            #pragma unroll
            for (int t = 0; t < 9; ++t) acc[t][v] *= sc;
            s[0][v] = exp2f(s[0][v] - mn);
            s[1][v] = exp2f(s[1][v] - mn);
        }

        // ---- transpose P (C layout -> A layout) through per-wave LDS ----
        __bf16* pw = pbuf[wave];
        #pragma unroll
        for (int nt = 0; nt < 2; ++nt) {
            #pragma unroll
            for (int v = 0; v < 8; ++v)
                pw[(v + 8 * half) * 32 + nt * 16 + l16] = (__bf16)s[nt][v];
        }
        asm volatile("s_wait_dscnt 0" ::: "memory");    // same-wave LDS RAW fence
        union { v16bf v; v8bf h[2]; } pf;
        const __bf16* pr = pw + l16 * 32;
        pf.h[0] = *(const v8bf*)(pr + 8 * half);
        pf.h[1] = *(const v8bf*)(pr + 16 + 8 * half);

        // ---- O|rowsum += P @ [V|1] : 9 WMMAs (tile 8 = ones block) ----
        v16bf bv[9];
        #pragma unroll
        for (int t = 0; t < 9; ++t) {
            union { v16bf v; v8bf h[2]; } b;
            const __bf16* vp = (t < 8) ? &ldsKT[cur][t * 16 + l16][8 * half] : &ldsV[cur][l16][128];
            b.h[0] = *(const v8bf*)(vp);
            b.h[1] = *(const v8bf*)(vp + (t < 8 ? 16 : 8));
            bv[t] = b.v;
        }
        #pragma unroll
        for (int t = 0; t < 9; ++t)
            acc[t] = __builtin_amdgcn_wmma_f32_16x16x32_bf16(
                         false, pf.v, false, bv[t], (short)0, acc[t], false, false);
    }

    // ---- epilogue: normalize by matrix-pipe rowsum, add lang_emb, store ----
    #pragma unroll
    for (int v = 0; v < 8; ++v) {
        const float inv = 1.0f / acc[8][v];             // softmax denominator
        const size_t row = (size_t)(rowbase + v + 8 * half);
        #pragma unroll
        for (int t = 0; t < 8; ++t) {
            const size_t idx = row * DIM + t * 16 + l16;
            token[idx] = acc[t][v] * inv + lang[idx];
        }
    }
}

// ---------------------------------------------------------------------------
// Kernel 3: output gather. One wave per token; lane moves a float4.
// ---------------------------------------------------------------------------
__global__ void gather_kernel(const int*   __restrict__ x,
                              const float* __restrict__ special,
                              const float* __restrict__ token,
                              float* __restrict__ out) {
    int row  = blockIdx.x * (blockDim.x >> 5) + (threadIdx.x >> 5);
    int lane = threadIdx.x & 31;
    if (row >= BT) return;
    int idx = x[row];
    const v4f* src = (idx < NSPECIAL)
        ? (const v4f*)(special + (size_t)idx * DIM)
        : (const v4f*)(token + (size_t)(idx - NSPECIAL) * DIM);
    ((v4f*)(out + (size_t)row * DIM))[lane] = src[lane];
}

// ---------------------------------------------------------------------------
extern "C" void kernel_launch(void* const* d_in, const int* in_sizes, int n_in,
                              void* d_out, int out_size, void* d_ws, size_t ws_size,
                              hipStream_t stream) {
    const float* ngram_emb = (const float*)d_in[0];   // [32001,128]
    const float* latent    = (const float*)d_in[1];   // [10000,128]
    const float* special   = (const float*)d_in[2];   // [4,128]
    const int*   ngram_ids = (const int*)  d_in[3];   // [32000*26]
    const int*   offsets   = (const int*)  d_in[4];   // [32000]
    const int*   x         = (const int*)  d_in[5];   // [8,1024]
    const int n_total = in_sizes[3];

    // workspace layout (~46.5 MB)
    char* ws = (char*)d_ws;
    size_t off = 0;
    auto take = [&](size_t bytes) {
        void* p = ws + off;
        off = (off + bytes + 255) & ~(size_t)255;
        return p;
    };
    float*  lang  = (float*) take(sizeof(float)  * (size_t)V_WORD * DIM);
    float*  token = (float*) take(sizeof(float)  * (size_t)V_WORD * DIM);
    __bf16* qb    = (__bf16*)take(sizeof(__bf16) * (size_t)V_WORD * DIM);
    __bf16* lat   = (__bf16*)take(sizeof(__bf16) * (size_t)LATENT_PAD * DIM);
    __bf16* latT  = (__bf16*)take(sizeof(__bf16) * (size_t)DIM * LATENT_PAD);

    prep_latent_kernel<<<(LATENT_PAD * DIM + 255) / 256, 256, 0, stream>>>(latent, lat, latT);
    bag_tanh_kernel<<<(V_WORD + 7) / 8, 256, 0, stream>>>(ngram_emb, ngram_ids, offsets,
                                                          n_total, lang, qb);
    latent_attn_kernel<<<V_WORD / 128, 256, 0, stream>>>(qb, lat, latT, lang, token);
    gather_kernel<<<(BT + 7) / 8, 256, 0, stream>>>(x, special, token, (float*)d_out);
}
